// RotaryAttention_37735582662740
// MI455X (gfx1250) — compile-verified, matrix operand order checked by reading, not run on hardware
//
#include <hip/hip_runtime.h>
#include <hip/hip_bf16.h>

// ---------------------------------------------------------------------------
// RotaryAttention forward for MI455X (gfx1250), wave32 + WMMA bf16 + TDM.
//   B=4, S=2048, HIDDEN=1024, H=16, D=64
// Pipeline: fp32->bf16 pre-pass -> qkv GEMM (+RoPE) -> flash attn -> out-proj.
// GEMM tiles are DMA'd to LDS by the Tensor Data Mover (tensor_load_to_lds,
// TENSORcnt), matmuls on v_wmma_f32_16x16x32_bf16, cross-lane ops on DPP.
// ---------------------------------------------------------------------------

typedef __attribute__((ext_vector_type(16))) __bf16        v16bf;
typedef __attribute__((ext_vector_type(8)))  float         v8f;
typedef __attribute__((ext_vector_type(4)))  unsigned int  v4u;
typedef __attribute__((ext_vector_type(8)))  int           v8i_t;
typedef __attribute__((ext_vector_type(4)))  int           v4i_t;

#define HIDDEN 1024
#define NHEAD  16
#define HDIM   64
#define SEQ    2048
#define BATCH  4

struct bf16frag_bits { uint4 lo, hi; };

__device__ __forceinline__ unsigned short f2bf(float f) {
  unsigned int u = __builtin_bit_cast(unsigned int, f);
  unsigned int r = u + 0x7FFFu + ((u >> 16) & 1u);   // round-to-nearest-even
  return (unsigned short)(r >> 16);
}

// --- DPP cross-lane helpers (constant patterns -> VALU, no LDS crossbar) ---
template <int CTRL>
__device__ __forceinline__ float dppf(float v) {
  return __builtin_bit_cast(float,
      __builtin_amdgcn_mov_dpp(__builtin_bit_cast(int, v), CTRL,
                               0xF, 0xF, /*bound_ctrl=*/true));
}
#define DPP_XOR1   0xB1   // quad_perm(1,0,3,2)
#define DPP_XOR2   0x4E   // quad_perm(2,3,0,1)
#define DPP_MIRR8  0x141  // row_half_mirror
#define DPP_MIRR16 0x140  // row_mirror (within 16-lane row)

__device__ __forceinline__ float group16_max(float v) {
  v = fmaxf(v, dppf<DPP_XOR1>(v));
  v = fmaxf(v, dppf<DPP_XOR2>(v));
  v = fmaxf(v, dppf<DPP_MIRR8>(v));
  v = fmaxf(v, dppf<DPP_MIRR16>(v));
  return v;
}
__device__ __forceinline__ float group16_sum(float v) {
  v += dppf<DPP_XOR1>(v);
  v += dppf<DPP_XOR2>(v);
  v += dppf<DPP_MIRR8>(v);
  v += dppf<DPP_MIRR16>(v);
  return v;
}

// A/B fragment convention for v_wmma_f32_16x16x32_bf16 (per ISA 7.12.2):
// lane L: row = base + (L&15); halfs 0..7  = K[(L>>4)*8 .. +8)
//                              halfs 8..15 = K[16 + (L>>4)*8 .. +8)
__device__ __forceinline__ v16bf load_frag_lds(const unsigned short* base,
                                               int stride, int lane) {
  const unsigned short* p = base + (lane & 15) * stride + ((lane >> 4) << 3);
  bf16frag_bits f;
  f.lo = *(const uint4*)p;
  f.hi = *(const uint4*)(p + 16);
  return __builtin_bit_cast(v16bf, f);
}

__device__ __forceinline__ v16bf load_frag_global(const unsigned short* base,
                                                  int stride, int lane) {
  const unsigned short* p = base + (long long)(lane & 15) * stride + ((lane >> 4) << 3);
  bf16frag_bits f;
  f.lo = *(const uint4*)p;
  f.hi = *(const uint4*)(p + 16);
  return __builtin_bit_cast(v16bf, f);
}

__device__ __forceinline__ v8f wmma_bf16(v16bf a, v16bf b, v8f c) {
  return __builtin_amdgcn_wmma_f32_16x16x32_bf16(
      false, a, false, b, (short)0, c, false, false);
}

// --- Tensor Data Mover: 2D bf16 tile -> LDS with padded row pitch ----------
// D# per ISA ch.8: group0 = {flags/count, lds_addr, global_addr lo/hi+type},
// group1 = {mask/size/pad, tensor dims, tile dims, strides}. data_size=1 (2B).
// pad_interval=3 (every 16 DWORDs = one 32-elem row), pad_amount=3 (4 DWORDs)
// -> LDS row pitch 80B == LDSTRIDE(40) shorts.
__device__ __forceinline__ unsigned lds_addr_of(void* p) {
  return (unsigned)(unsigned long long)((__attribute__((address_space(3))) char*)p);
}

__device__ __forceinline__ void tdm_load_tile_2d(
    unsigned lds_addr, const unsigned short* gptr,
    unsigned tensor_d0, unsigned tensor_d1, unsigned d0_stride,
    unsigned tile_d0, unsigned tile_d1,
    unsigned pad_interval, unsigned pad_amount, bool pad_en) {
  unsigned long long ga = (unsigned long long)gptr;
  v4u g0;
  g0.x = 1u;                                         // count=1, user descriptor
  g0.y = lds_addr;                                   // lds_addr [63:32]
  g0.z = (unsigned)(ga & 0xffffffffu);               // global_addr [95:64]
  g0.w = (unsigned)((ga >> 32) & 0x1ffffffu) | (2u << 30);  // addr hi + type=2
  v8i_t g1;
  g1[0] = (int)((1u << 16) |                         // data_size = 2 bytes
                (pad_en ? ((1u << 20) | (pad_interval << 22) | (pad_amount << 25))
                        : 0u));
  g1[1] = (int)((tensor_d0 & 0xffffu) << 16);                        // dim0 lo
  g1[2] = (int)(((tensor_d0 >> 16) & 0xffffu) | ((tensor_d1 & 0xffffu) << 16));
  g1[3] = (int)(((tensor_d1 >> 16) & 0xffffu) | ((tile_d0 & 0xffffu) << 16));
  g1[4] = (int)(tile_d1 & 0xffffu);                  // tile_dim1 (tile_dim2=0)
  g1[5] = (int)d0_stride;                            // dim0_stride lo32
  g1[6] = 0;                                         // stride hi / dim1_stride
  g1[7] = 0;
  v4i_t z4 = {0, 0, 0, 0};
#if defined(__clang_major__) && (__clang_major__ >= 23)
  v8i_t z8 = {0, 0, 0, 0, 0, 0, 0, 0};
  __builtin_amdgcn_tensor_load_to_lds(g0, g1, z4, z4, z8, 0);
#else
  __builtin_amdgcn_tensor_load_to_lds(g0, g1, z4, z4, 0);
#endif
}

// ---------------------------------------------------------------------------
// Kernel 0: fp32 -> bf16 conversion pre-pass (bandwidth trivial: ~50MB total).
// ---------------------------------------------------------------------------
__global__ __launch_bounds__(256) void cvt_f32_bf16(
    const float* __restrict__ src, unsigned short* __restrict__ dst, int n4) {
  int i = blockIdx.x * 256 + threadIdx.x;
  if (i < n4) {
    float4 f = ((const float4*)src)[i];
    unsigned lo = (unsigned)f2bf(f.x) | ((unsigned)f2bf(f.y) << 16);
    unsigned hi = (unsigned)f2bf(f.z) | ((unsigned)f2bf(f.w) << 16);
    ((uint2*)dst)[i] = make_uint2(lo, hi);
  }
}

// ---------------------------------------------------------------------------
// Kernel 1: qkv = x @ w_qkv^T  (M=8192, N=3072, K=1024) on bf16 inputs.
// Tiles 128x128, K-step 32; A/B tiles DMA'd by TDM into double-buffered LDS
// (wave 0 issues, TENSORcnt + barrier synchronizes). Epilogue applies RoPE
// (DPP pair swap) and scatters bf16 q/k/v into [B,H,S,D].
// ---------------------------------------------------------------------------
#define LDSTRIDE 40   // shorts; 80B row pitch (TDM pad) -> aligned b128

__global__ __launch_bounds__(256) void qkv_gemm_rope(
    const unsigned short* __restrict__ xb, const unsigned short* __restrict__ wb,
    unsigned short* __restrict__ qb, unsigned short* __restrict__ kb,
    unsigned short* __restrict__ vb) {
  __shared__ unsigned short As[2][128 * LDSTRIDE];
  __shared__ unsigned short Bs[2][128 * LDSTRIDE];

  const int tid  = threadIdx.x;
  const int lane = tid & 31;
  const int wid  = tid >> 5;
  const int m0   = blockIdx.x * 128;          // 64 tiles
  const int n0   = blockIdx.y * 128;          // 24 tiles
  const int wm   = wid >> 2;                  // 0..1 -> 64 rows
  const int wn   = wid & 3;                   // 0..3 -> 32 cols

  v8f c[4][2] = {};

  auto issue = [&](int buf, int kt) {
    tdm_load_tile_2d(lds_addr_of(&As[buf][0]),
                     xb + (long long)m0 * 1024 + kt,
                     1024, 8192, 1024, 32, 128, 3, 3, true);
    tdm_load_tile_2d(lds_addr_of(&Bs[buf][0]),
                     wb + (long long)n0 * 1024 + kt,
                     1024, 3072, 1024, 32, 128, 3, 3, true);
  };
  auto compute = [&](const unsigned short* Ab, const unsigned short* Bb) {
    v16bf a[4], b[2];
#pragma unroll
    for (int i = 0; i < 4; ++i)
      a[i] = load_frag_lds(Ab + (wm * 64 + i * 16) * LDSTRIDE, LDSTRIDE, lane);
#pragma unroll
    for (int j = 0; j < 2; ++j)
      b[j] = load_frag_lds(Bb + (wn * 32 + j * 16) * LDSTRIDE, LDSTRIDE, lane);
#pragma unroll
    for (int i = 0; i < 4; ++i)
#pragma unroll
      for (int j = 0; j < 2; ++j) c[i][j] = wmma_bf16(a[i], b[j], c[i][j]);
  };

  if (wid == 0) {
    issue(0, 0);
    __builtin_amdgcn_s_wait_tensorcnt(0);
  }
  __syncthreads();
  int buf = 0;
  for (int kt = 0; kt < 1024; kt += 32, buf ^= 1) {
    if (wid == 0 && kt + 32 < 1024) issue(buf ^ 1, kt + 32);  // DMA overlaps WMMA
    compute(&As[buf][0], &Bs[buf][0]);
    if (wid == 0) __builtin_amdgcn_s_wait_tensorcnt(0);
    __syncthreads();
  }

  // Epilogue: RoPE for q/k, scatter bf16 into [B,H,S,D]. Within one 16-wide
  // fragment the (h, q/k/v) selector is wave-uniform, so no divergence.
  const int lrow = lane & 15;
  const int hi16 = lane >> 4;
#pragma unroll
  for (int i = 0; i < 4; ++i) {
#pragma unroll
    for (int j = 0; j < 2; ++j) {
      int n     = n0 + wn * 32 + j * 16 + lrow;
      int h     = n / 192;
      int rr    = n % 192;
      int which = rr >> 6;      // 0=q 1=k 2=v (uniform per fragment)
      int d     = rr & 63;
      unsigned short* dst = (which == 0) ? qb : (which == 1) ? kb : vb;
      // theta = 10000^(p/32) -> invtheta = exp(-p * ln(10000)/32), p = d>>1
      float invtheta = __expf((float)(d >> 1) * -0.28782313662425572f);
      long long hb = ((long long)h * SEQ) * HDIM;
#pragma unroll
      for (int r = 0; r < 8; ++r) {
        int m    = m0 + wm * 64 + i * 16 + (hi16 ? r + 8 : r);
        int bidx = m >> 11;
        int s    = m & 2047;
        float val = c[i][j][r];
        if (which < 2) {
          float partner = dppf<DPP_XOR1>(val);   // value at d^1 (same row)
          float ang = (float)s * invtheta;
          float sn  = __sinf(ang);
          float cs  = __cosf(ang);
          val = (d & 1) ? (val * cs + partner * sn)    // odd slot
                        : (val * cs - partner * sn);   // even slot
        }
        long long dsti = (long long)bidx * NHEAD * SEQ * HDIM + hb +
                         (long long)s * HDIM + d;
        dst[dsti] = f2bf(val);
      }
    }
  }
}

// ---------------------------------------------------------------------------
// Kernel 2: flash attention. One block per (b*H, 64-row q tile).
// 128 threads = 4 waves; wave owns 16 q-rows x full 64-key tiles.
// Q/K fragments load straight from global; V transposed through LDS (raw tile
// prefetched one iteration ahead); P re-shapes C->A layout via per-wave LDS.
// ---------------------------------------------------------------------------
#define VSTRIDE 72   // shorts; 144B pitch, 16B aligned

__global__ __launch_bounds__(128) void flash_attn(
    const unsigned short* __restrict__ qb, const unsigned short* __restrict__ kb,
    const unsigned short* __restrict__ vb, unsigned short* __restrict__ ob) {
  __shared__ unsigned short Vt[64 * VSTRIDE];        // V^T tile: [d][key]
  __shared__ unsigned short Ps[4 * 16 * VSTRIDE];    // per-wave P staging

  const int tid  = threadIdx.x;
  const int lane = tid & 31;
  const int wid  = tid >> 5;
  const int qt   = blockIdx.x & 31;        // 32 q-tiles of 64
  const int bh   = blockIdx.x >> 5;        // b*H + h
  const long long base = (long long)bh * SEQ * HDIM;
  const int q0   = qt * 64 + wid * 16;

  v16bf aq0 = load_frag_global(qb + base + (long long)q0 * HDIM + 0,  HDIM, lane);
  v16bf aq1 = load_frag_global(qb + base + (long long)q0 * HDIM + 32, HDIM, lane);

  v8f acc[4] = {};
  float m_s[8], l_s[8];
#pragma unroll
  for (int r = 0; r < 8; ++r) { m_s[r] = -1e30f; l_s[r] = 0.0f; }

  unsigned short* pw = &Ps[wid * 16 * VSTRIDE];

  uint4 vr[4];   // V-tile raw prefetch (4 chunks of 8 halfs per thread)
  auto loadV = [&](int kt) {
#pragma unroll
    for (int i = 0; i < 4; ++i) {
      int c8  = tid + i * 128;             // 512 chunks of 8 halfs
      int key = c8 >> 3;
      int d0  = (c8 & 7) * 8;
      vr[i] = *(const uint4*)(vb + base + (long long)(kt + key) * HDIM + d0);
    }
  };
  auto storeV = [&]() {
#pragma unroll
    for (int i = 0; i < 4; ++i) {
      int c8  = tid + i * 128;
      int key = c8 >> 3;
      int d0  = (c8 & 7) * 8;
      Vt[(d0 + 0) * VSTRIDE + key] = (unsigned short)(vr[i].x & 0xffff);
      Vt[(d0 + 1) * VSTRIDE + key] = (unsigned short)(vr[i].x >> 16);
      Vt[(d0 + 2) * VSTRIDE + key] = (unsigned short)(vr[i].y & 0xffff);
      Vt[(d0 + 3) * VSTRIDE + key] = (unsigned short)(vr[i].y >> 16);
      Vt[(d0 + 4) * VSTRIDE + key] = (unsigned short)(vr[i].z & 0xffff);
      Vt[(d0 + 5) * VSTRIDE + key] = (unsigned short)(vr[i].z >> 16);
      Vt[(d0 + 6) * VSTRIDE + key] = (unsigned short)(vr[i].w & 0xffff);
      Vt[(d0 + 7) * VSTRIDE + key] = (unsigned short)(vr[i].w >> 16);
    }
  };

  loadV(0);

  for (int kt = 0; kt < SEQ; kt += 64) {
    __syncthreads();   // previous Vt consumers done
    storeV();          // commit this tile's V^T
    if (kt + 64 < SEQ) loadV(kt + 64);   // prefetch next, overlaps compute
    __syncthreads();

    // --- scores S = Q K^T  (16 x 64 per wave) ---
    v8f cs[4];
#pragma unroll
    for (int j = 0; j < 4; ++j) {
      const unsigned short* kp = kb + base + (long long)(kt + j * 16) * HDIM;
      v16bf b0 = load_frag_global(kp + 0,  HDIM, lane);
      v16bf b1 = load_frag_global(kp + 32, HDIM, lane);
      v8f z = {};
      z = wmma_bf16(aq0, b0, z);
      cs[j] = wmma_bf16(aq1, b1, z);
    }

    // --- online softmax (row = (lane<16 ? r : r+8); DPP row reductions) ---
    float scale[8];
#pragma unroll
    for (int r = 0; r < 8; ++r) {
      float v = fmaxf(fmaxf(cs[0][r], cs[1][r]), fmaxf(cs[2][r], cs[3][r]));
      v = group16_max(v);
      float mnew = fmaxf(m_s[r], v);
      scale[r] = __expf(m_s[r] - mnew);
      m_s[r] = mnew;
      l_s[r] *= scale[r];
    }
#pragma unroll
    for (int f = 0; f < 4; ++f)
#pragma unroll
      for (int r = 0; r < 8; ++r) acc[f][r] *= scale[r];

    float psum[8];
#pragma unroll
    for (int r = 0; r < 8; ++r) psum[r] = 0.0f;
#pragma unroll
    for (int j = 0; j < 4; ++j) {
#pragma unroll
      for (int r = 0; r < 8; ++r) {
        float p = __expf(cs[j][r] - m_s[r]);
        psum[r] += p;
        int row = (lane < 16) ? r : r + 8;
        pw[row * VSTRIDE + j * 16 + (lane & 15)] = f2bf(p);   // C-layout scatter
      }
    }
#pragma unroll
    for (int r = 0; r < 8; ++r) l_s[r] += group16_sum(psum[r]);

    // P staging is per-wave private; wait on DS before re-reading as A frags.
    asm volatile("s_wait_dscnt 0" ::: "memory");

    v16bf ap0 = load_frag_lds(pw,      VSTRIDE, lane);   // keys  0..31
    v16bf ap1 = load_frag_lds(pw + 32, VSTRIDE, lane);   // keys 32..63
#pragma unroll
    for (int f = 0; f < 4; ++f) {
      v16bf bv0 = load_frag_lds(&Vt[(f * 16) * VSTRIDE],      VSTRIDE, lane);
      v16bf bv1 = load_frag_lds(&Vt[(f * 16) * VSTRIDE + 32], VSTRIDE, lane);
      acc[f] = wmma_bf16(ap0, bv0, acc[f]);
      acc[f] = wmma_bf16(ap1, bv1, acc[f]);
    }
  }

  // --- normalize and write o[b,h,q,d] as bf16 ---
#pragma unroll
  for (int f = 0; f < 4; ++f) {
#pragma unroll
    for (int r = 0; r < 8; ++r) {
      int row = q0 + ((lane < 16) ? r : r + 8);
      int d   = f * 16 + (lane & 15);
      float o = acc[f][r] / l_s[r];
      ob[base + (long long)row * HDIM + d] = f2bf(o);
    }
  }
}

// ---------------------------------------------------------------------------
// Kernel 3: out = attn_flat @ w_out^T  (M=8192, N=1024, K=1024), fp32 out.
// A comes from the head-major bf16 attention buffer; a 32-wide K step stays
// inside one head (h = kt>>6 fixed per step), so the A tile is a plain 2D
// TDM tile with row stride HDIM. Same TDM double buffering as kernel 1.
// ---------------------------------------------------------------------------
__global__ __launch_bounds__(256) void out_proj(
    const unsigned short* __restrict__ ab, const unsigned short* __restrict__ wb,
    float* __restrict__ out) {
  __shared__ unsigned short As[2][128 * LDSTRIDE];
  __shared__ unsigned short Bs[2][128 * LDSTRIDE];

  const int tid  = threadIdx.x;
  const int lane = tid & 31;
  const int wid  = tid >> 5;
  const int m0   = blockIdx.x * 128;   // 64 tiles (all rows share one b)
  const int n0   = blockIdx.y * 128;   // 8 tiles
  const int wm   = wid >> 2;
  const int wn   = wid & 3;
  const int bidx = m0 >> 11;
  const int s0   = m0 & 2047;

  v8f c[4][2] = {};

  auto issue = [&](int buf, int kt) {
    int h = kt >> 6, d0 = kt & 63;
    tdm_load_tile_2d(lds_addr_of(&As[buf][0]),
                     ab + (((long long)(bidx * NHEAD + h)) * SEQ + s0) * HDIM + d0,
                     HDIM, SEQ, HDIM, 32, 128, 3, 3, true);
    tdm_load_tile_2d(lds_addr_of(&Bs[buf][0]),
                     wb + (long long)n0 * 1024 + kt,
                     1024, 1024, 1024, 32, 128, 3, 3, true);
  };
  auto compute = [&](const unsigned short* Ab, const unsigned short* Bb) {
    v16bf a[4], b[2];
#pragma unroll
    for (int i = 0; i < 4; ++i)
      a[i] = load_frag_lds(Ab + (wm * 64 + i * 16) * LDSTRIDE, LDSTRIDE, lane);
#pragma unroll
    for (int j = 0; j < 2; ++j)
      b[j] = load_frag_lds(Bb + (wn * 32 + j * 16) * LDSTRIDE, LDSTRIDE, lane);
#pragma unroll
    for (int i = 0; i < 4; ++i)
#pragma unroll
      for (int j = 0; j < 2; ++j) c[i][j] = wmma_bf16(a[i], b[j], c[i][j]);
  };

  if (wid == 0) {
    issue(0, 0);
    __builtin_amdgcn_s_wait_tensorcnt(0);
  }
  __syncthreads();
  int buf = 0;
  for (int kt = 0; kt < 1024; kt += 32, buf ^= 1) {
    if (wid == 0 && kt + 32 < 1024) issue(buf ^ 1, kt + 32);
    compute(&As[buf][0], &Bs[buf][0]);
    if (wid == 0) __builtin_amdgcn_s_wait_tensorcnt(0);
    __syncthreads();
  }

  const int lrow = lane & 15;
  const int hi16 = lane >> 4;
#pragma unroll
  for (int i = 0; i < 4; ++i)
#pragma unroll
    for (int j = 0; j < 2; ++j) {
      int n = n0 + wn * 32 + j * 16 + lrow;
#pragma unroll
      for (int r = 0; r < 8; ++r) {
        int m = m0 + wm * 64 + i * 16 + (hi16 ? r + 8 : r);
        out[(long long)m * 1024 + n] = c[i][j][r];
      }
    }
}

// ---------------------------------------------------------------------------
extern "C" void kernel_launch(void* const* d_in, const int* in_sizes, int n_in,
                              void* d_out, int out_size, void* d_ws, size_t ws_size,
                              hipStream_t stream) {
  const float* x     = (const float*)d_in[0];   // [4,2048,1024]
  const float* w_qkv = (const float*)d_in[1];   // [3072,1024]
  const float* w_out = (const float*)d_in[2];   // [1024,1024]
  float* out = (float*)d_out;                   // [4,2048,1024]

  const long long X_ELEMS   = (long long)BATCH * SEQ * HIDDEN;       // 8388608
  const long long WQKV_EL   = 3LL * HIDDEN * HIDDEN;                 // 3145728
  const long long WOUT_EL   = (long long)HIDDEN * HIDDEN;            // 1048576
  const long long QKV_ELEMS = (long long)BATCH * NHEAD * SEQ * HDIM; // 8388608

  unsigned short* ws  = (unsigned short*)d_ws;   // ~92 MB bf16 scratch
  unsigned short* xbb = ws;
  unsigned short* wqb = xbb + X_ELEMS;
  unsigned short* wob = wqb + WQKV_EL;
  unsigned short* qb  = wob + WOUT_EL;
  unsigned short* kb  = qb + QKV_ELEMS;
  unsigned short* vb  = kb + QKV_ELEMS;
  unsigned short* ob  = vb + QKV_ELEMS;

  // fp32 -> bf16 pre-pass (one-shot, bandwidth-trivial)
  cvt_f32_bf16<<<(int)(X_ELEMS / 4 + 255) / 256, 256, 0, stream>>>(x, xbb, (int)(X_ELEMS / 4));
  cvt_f32_bf16<<<(int)(WQKV_EL / 4 + 255) / 256, 256, 0, stream>>>(w_qkv, wqb, (int)(WQKV_EL / 4));
  cvt_f32_bf16<<<(int)(WOUT_EL / 4 + 255) / 256, 256, 0, stream>>>(w_out, wob, (int)(WOUT_EL / 4));

  dim3 g1(64, 24);   // 8192/128 x 3072/128
  qkv_gemm_rope<<<g1, 256, 0, stream>>>(xbb, wqb, qb, kb, vb);

  flash_attn<<<BATCH * NHEAD * (SEQ / 64), 128, 0, stream>>>(qb, kb, vb, ob);

  dim3 g2(64, 8);    // 8192/128 x 1024/128
  out_proj<<<g2, 256, 0, stream>>>(ob, wob, out);
}
